// GCN_Danger_31361851195410
// MI455X (gfx1250) — matrix operand order repaired and checked by reading, not compiled, not run on hardware
//
#include <hip/hip_runtime.h>

// ---------------------------------------------------------------------------
// GCN (3x GCNConv + global_max_pool + MLP head) for MI455X / gfx1250.
// GEMMs use v_wmma_f32_16x16x32_bf16 (bf16 in, fp32 accum); weights are
// pre-transposed/pre-converted to bf16 once per layer so the GEMM hot loop is
// pure b128 LDS traffic + WMMA.  Edge aggregation gathers bf16 features
// (half the HBM gather traffic) and scatter-adds fp32 in L2.
// ---------------------------------------------------------------------------

typedef __attribute__((ext_vector_type(16))) __bf16 v16bf;
typedef __attribute__((ext_vector_type(8)))  __bf16 bf16x8;
typedef __attribute__((ext_vector_type(8)))  float  v8f;

static __device__ __forceinline__ unsigned short f2bf_raw(float f) {
  union { float f; unsigned u; } v; v.f = f;
  unsigned r = v.u + 0x7FFFu + ((v.u >> 16) & 1u);   // round-to-nearest-even
  return (unsigned short)(r >> 16);
}
static __device__ __forceinline__ float bf2f_raw(unsigned short h) {
  union { unsigned u; float f; } v; v.u = ((unsigned)h) << 16;
  return v.f;
}

// ------------------------------- utilities ---------------------------------
__global__ void k_zero_f32(float* p, long long n) {
  long long i = (long long)blockIdx.x * blockDim.x + threadIdx.x;
  long long s = (long long)gridDim.x * blockDim.x;
  for (; i < n; i += s) p[i] = 0.0f;
}

__global__ void k_deg(const int* __restrict__ dst, float* __restrict__ deg, int E) {
  int i = blockIdx.x * blockDim.x + threadIdx.x;
  int s = gridDim.x * blockDim.x;
  for (; i < E; i += s) atomicAdd(&deg[dst[i]], 1.0f);
}

__global__ void k_rsqrt(const float* __restrict__ deg, float* __restrict__ dis, int N) {
  int i = blockIdx.x * blockDim.x + threadIdx.x;
  if (i < N) dis[i] = rsqrtf(deg[i] + 1.0f);
}

__global__ void k_norm(const int* __restrict__ src, const int* __restrict__ dst,
                       const float* __restrict__ dis, float* __restrict__ nrm, int E) {
  int i = blockIdx.x * blockDim.x + threadIdx.x;
  int s = gridDim.x * blockDim.x;
  for (; i < E; i += s) nrm[i] = dis[src[i]] * dis[dst[i]];
}

// W[K x OUT] (fp32, row-major) -> WT[OUT x Kp] (bf16, transposed, zero-padded)
__global__ void k_prep_wt(const float* __restrict__ W, int K, int Kp, int OUT,
                          unsigned short* __restrict__ WT) {
  int idx = blockIdx.x * blockDim.x + threadIdx.x;
  int total = OUT * Kp;
  if (idx >= total) return;
  int o = idx / Kp, k = idx % Kp;
  WT[idx] = (k < K) ? f2bf_raw(W[(long long)k * OUT + o]) : (unsigned short)0;
}

// --------------------------- WMMA tiled GEMM -------------------------------
// Y[N x OUT] (bf16) = Xin[N x K] @ W[K x OUT].  Block = 128 thr = 4 waves,
// each wave owns a 16-row strip; K traversed in steps of 32 (WMMA K).
// WT is the pre-transposed bf16 weight: WT[o * Kp + k].
template <int OUT, bool INF32>
__global__ void k_gemm_wmma(const void* __restrict__ Xin, int K, int Kp, int N,
                            const unsigned short* __restrict__ WT,
                            unsigned short* __restrict__ Y) {
  __shared__ __attribute__((aligned(16))) unsigned short Alds[4][16][32];   // [wave][m][k]
  __shared__ __attribute__((aligned(16))) unsigned short Wlds[OUT][32];     // [o][k]

  const int tid  = threadIdx.x;
  const int lane = tid & 31;
  const int wv   = tid >> 5;
  const int m0   = blockIdx.x * 64 + wv * 16;

  constexpr int NT = OUT / 16;
  v8f acc[NT];
#pragma unroll
  for (int t = 0; t < NT; ++t)
#pragma unroll
    for (int r = 0; r < 8; ++r) acc[t][r] = 0.0f;

  const float*          Xf = (const float*)Xin;
  const unsigned short* Xb = (const unsigned short*)Xin;

  // A-stage assignment: 2 lanes per row, each copies a contiguous 16-col half.
  const int rA   = lane >> 1;
  const int chA  = (lane & 1) * 16;
  const int growA = m0 + rA;

  const int mA  = lane & 15;          // A-frag row
  const int kbA = (lane >> 4) * 8;    // A-frag K base (ISA 16-bit A 16x32 layout)
  const int nB  = lane & 15;          // B-frag column
  const int kbB = (lane >> 4) * 8;    // B-frag K base (same K order as A frag)

  for (int ko = 0; ko < Kp; ko += 32) {
    // ---- stage B tile: pure 64B copy per thread (no conversion) ----
    if (tid < OUT) {
      const uint4* sp = (const uint4*)(WT + (long long)tid * Kp + ko);
      uint4* dp = (uint4*)&Wlds[tid][0];
      dp[0] = sp[0]; dp[1] = sp[1]; dp[2] = sp[2]; dp[3] = sp[3];
    }
    // ---- stage A tile: 32B contiguous per lane ----
    {
      uint4* dp = (uint4*)&Alds[wv][rA][chA];
      if (INF32) {
        if (growA < N && ko + 32 <= K) {
          // fast path: 64B of fp32, convert to 32B of bf16
          float tf[16];
          __builtin_memcpy(tf, Xf + (long long)growA * K + ko + chA, 64);
          unsigned short tmp[16];
#pragma unroll
          for (int i = 0; i < 16; ++i) tmp[i] = f2bf_raw(tf[i]);
          __builtin_memcpy(dp, tmp, 32);
        } else {
          unsigned short tmp[16];
#pragma unroll
          for (int i = 0; i < 16; ++i) {
            int gk = ko + chA + i;
            tmp[i] = (growA < N && gk < K)
                         ? f2bf_raw(Xf[(long long)growA * K + gk]) : (unsigned short)0;
          }
          __builtin_memcpy(dp, tmp, 32);
        }
        if (ko + 32 < Kp && growA < N)
          __builtin_prefetch(Xf + (long long)growA * K + ko + 32 + chA, 0, 1);
      } else {
        if (growA < N) {
          const uint4* sp = (const uint4*)(Xb + (long long)growA * K + ko + chA);
          dp[0] = sp[0]; dp[1] = sp[1];
          if (ko + 32 < Kp)
            __builtin_prefetch(Xb + (long long)growA * K + ko + 32 + chA, 0, 1);
        } else {
          uint4 z = {0u, 0u, 0u, 0u};
          dp[0] = z; dp[1] = z;
        }
      }
    }
    __syncthreads();

    // A fragment: A[m][kbA..kbA+7] ++ A[m][kbA+16..kbA+23]  (two b128 LDS loads)
    bf16x8 alo, ahi;
    __builtin_memcpy(&alo, &Alds[wv][mA][kbA],      16);
    __builtin_memcpy(&ahi, &Alds[wv][mA][kbA + 16], 16);
    v16bf af;
#pragma unroll
    for (int i = 0; i < 8; ++i) { af[i] = alo[i]; af[8 + i] = ahi[i]; }

#pragma unroll
    for (int t = 0; t < NT; ++t) {
      // B fragment: W[kbB..kbB+7] ++ W[kbB+16..kbB+23] at column t*16+nB (matches A)
      bf16x8 blo, bhi;
      __builtin_memcpy(&blo, &Wlds[t * 16 + nB][kbB],     16);
      __builtin_memcpy(&bhi, &Wlds[t * 16 + nB][kbB + 16], 16);
      v16bf bf;
#pragma unroll
      for (int i = 0; i < 8; ++i) { bf[i] = blo[i]; bf[8 + i] = bhi[i]; }

      acc[t] = __builtin_amdgcn_wmma_f32_16x16x32_bf16(
          /*neg_a=*/false, af, /*neg_b=*/false, bf,
          /*c_mod=*/(short)0, acc[t], /*reuse_a=*/false, /*reuse_b=*/false);
    }
    __syncthreads();
  }

  // Store D per ISA C/D layout: row = (lane>>4)*8 + r, col = lane&15
  const int mD = (lane >> 4) * 8;
  const int nD = lane & 15;
#pragma unroll
  for (int t = 0; t < NT; ++t)
#pragma unroll
    for (int r = 0; r < 8; ++r) {
      int grow = m0 + mD + r;
      if (grow < N) Y[(long long)grow * OUT + t * 16 + nD] = f2bf_raw(acc[t][r]);
    }
}

// --------------------------- edge scatter-add ------------------------------
// One wave per edge: gather bf16 features of src, scale by norm, fp32
// atomic-add into dst row.  F = OUT/32 features per lane (one 8B/4B load).
template <int OUT>
__global__ void k_scatter(const unsigned short* __restrict__ xw,
                          const int* __restrict__ src, const int* __restrict__ dst,
                          const float* __restrict__ nrm,
                          float* __restrict__ agg, int E) {
  constexpr int F = OUT / 32;
  int gw   = (blockIdx.x * blockDim.x + threadIdx.x) >> 5;
  int lane = threadIdx.x & 31;
  int nw   = (gridDim.x * blockDim.x) >> 5;
  for (int e = gw; e < E; e += nw) {
    int s = src[e], d = dst[e];
    float nm = nrm[e];
    const unsigned short* xs = xw + (long long)s * OUT + lane * F;
    float* ad = agg + (long long)d * OUT + lane * F;
    unsigned short v[F];
    __builtin_memcpy(v, xs, F * 2);
#pragma unroll
    for (int j = 0; j < F; ++j) atomicAdd(&ad[j], bf2f_raw(v[j]) * nm);
  }
}

// ----------------- self-loop + bias + ReLU (-> bf16 or f32) ----------------
template <int OUT, bool F32OUT>
__global__ void k_finish(const float* __restrict__ agg, const unsigned short* __restrict__ xw,
                         const float* __restrict__ dis, const float* __restrict__ b,
                         void* __restrict__ hout, int N) {
  long long idx = (long long)blockIdx.x * blockDim.x + threadIdx.x;
  long long tot = (long long)N * OUT;
  long long s   = (long long)gridDim.x * blockDim.x;
  for (; idx < tot; idx += s) {
    int i = (int)(idx / OUT), f = (int)(idx % OUT);
    float dii = dis[i];
    float v = agg[idx] + bf2f_raw(xw[idx]) * dii * dii + b[f];
    v = fmaxf(v, 0.0f);
    if (F32OUT) ((float*)hout)[idx] = v;
    else        ((unsigned short*)hout)[idx] = f2bf_raw(v);
  }
}

// ------------------------------- max pool ----------------------------------
// ReLU outputs are >= 0, so int-bit atomicMax matches float max ordering.
__global__ void k_pool(const float* __restrict__ h3, const int* __restrict__ batch,
                       float* __restrict__ gmax, int N) {
  long long idx = (long long)blockIdx.x * blockDim.x + threadIdx.x;
  long long tot = (long long)N * 64;
  long long s   = (long long)gridDim.x * blockDim.x;
  for (; idx < tot; idx += s) {
    int i = (int)(idx >> 6), f = (int)(idx & 63);
    atomicMax((int*)&gmax[batch[i] * 64 + f], __float_as_int(h3[idx]));
  }
}

// ------------------------------- MLP head ----------------------------------
__global__ void k_head(const float* __restrict__ gmax,
                       const float* __restrict__ Wh1, const float* __restrict__ bh1,
                       const float* __restrict__ Wh2, const float* __restrict__ bh2,
                       float* __restrict__ out) {
  int g = threadIdx.x;
  if (g >= 64) return;
  float hid[32];
#pragma unroll
  for (int j = 0; j < 32; ++j) hid[j] = bh1[j];
  for (int f = 0; f < 64; ++f) {
    float x = gmax[g * 64 + f];
#pragma unroll
    for (int j = 0; j < 32; ++j) hid[j] += x * Wh1[f * 32 + j];
  }
  float o = bh2[0];
#pragma unroll
  for (int j = 0; j < 32; ++j) o += fmaxf(hid[j], 0.0f) * Wh2[j];
  out[g] = o;
}

// ---------------------------------------------------------------------------
extern "C" void kernel_launch(void* const* d_in, const int* in_sizes, int n_in,
                              void* d_out, int out_size, void* d_ws, size_t ws_size,
                              hipStream_t stream) {
  const float* x    = (const float*)d_in[0];
  const int*   ei   = (const int*)d_in[1];
  const int*   bat  = (const int*)d_in[2];
  const float* W1   = (const float*)d_in[3];
  const float* b1   = (const float*)d_in[4];
  const float* W2   = (const float*)d_in[5];
  const float* b2   = (const float*)d_in[6];
  const float* W3   = (const float*)d_in[7];
  const float* b3   = (const float*)d_in[8];
  const float* Wh1  = (const float*)d_in[9];
  const float* bh1  = (const float*)d_in[10];
  const float* Wh2  = (const float*)d_in[11];
  const float* bh2  = (const float*)d_in[12];

  const int N  = in_sizes[2];       // batch vector length = #nodes
  const int E  = in_sizes[1] / 2;   // edge_index is [2, E]
  const int K1 = in_sizes[0] / N;   // 397
  const int Kp1 = ((K1 + 31) / 32) * 32;
  const int* src = ei;
  const int* dst = ei + E;

  // workspace carve-up (~110 MB)
  char* ws = (char*)d_ws;
  auto au = [](size_t v) { return (v + 255) & ~(size_t)255; };
  size_t off = 0;
  float*          dis  = (float*)(ws + off);          off = au(off + (size_t)N * 4);
  float*          nrm  = (float*)(ws + off);          off = au(off + (size_t)E * 4);
  unsigned short* xw   = (unsigned short*)(ws + off); off = au(off + (size_t)N * 128 * 2);
  float*          agg  = (float*)(ws + off);          off = au(off + (size_t)N * 128 * 4);
  unsigned short* hbuf = (unsigned short*)(ws + off); off = au(off + (size_t)N * 128 * 2);
  float*          gmax = (float*)(ws + off);          off = au(off + (size_t)64 * 64 * 4);
  unsigned short* WT   = (unsigned short*)(ws + off); off = au(off + (size_t)128 * Kp1 * 2);
  float*          h3   = (float*)hbuf;   // layer-3 fp32 output reuses hbuf (same bytes)

  const int gemmBlocks = (N + 63) / 64;

  // degree -> deg^{-1/2} -> per-edge norm (agg[0..N) used as deg scratch)
  k_zero_f32<<<1024, 256, 0, stream>>>(agg, N);
  k_deg<<<2048, 256, 0, stream>>>(dst, agg, E);
  k_rsqrt<<<(N + 255) / 256, 256, 0, stream>>>(agg, dis, N);
  k_norm<<<2048, 256, 0, stream>>>(src, dst, dis, nrm, E);

  // ---- layer 1: x[N,397] @ W1 -> 128 ----
  k_prep_wt<<<(128 * Kp1 + 255) / 256, 256, 0, stream>>>(W1, K1, Kp1, 128, WT);
  k_gemm_wmma<128, true><<<gemmBlocks, 128, 0, stream>>>(x, K1, Kp1, N, WT, xw);
  k_zero_f32<<<4096, 256, 0, stream>>>(agg, (long long)N * 128);
  k_scatter<128><<<4096, 256, 0, stream>>>(xw, src, dst, nrm, agg, E);
  k_finish<128, false><<<4096, 256, 0, stream>>>(agg, xw, dis, b1, hbuf, N);

  // ---- layer 2: h[N,128] @ W2 -> 128 ----
  k_prep_wt<<<(128 * 128 + 255) / 256, 256, 0, stream>>>(W2, 128, 128, 128, WT);
  k_gemm_wmma<128, false><<<gemmBlocks, 128, 0, stream>>>(hbuf, 128, 128, N, WT, xw);
  k_zero_f32<<<4096, 256, 0, stream>>>(agg, (long long)N * 128);
  k_scatter<128><<<4096, 256, 0, stream>>>(xw, src, dst, nrm, agg, E);
  k_finish<128, false><<<4096, 256, 0, stream>>>(agg, xw, dis, b2, hbuf, N);

  // ---- layer 3: h[N,128] @ W3 -> 64 (fp32 output for pooling) ----
  k_prep_wt<<<(64 * 128 + 255) / 256, 256, 0, stream>>>(W3, 128, 128, 64, WT);
  k_gemm_wmma<64, false><<<gemmBlocks, 128, 0, stream>>>(hbuf, 128, 128, N, WT, xw);
  k_zero_f32<<<4096, 256, 0, stream>>>(agg, (long long)N * 64);
  k_scatter<64><<<4096, 256, 0, stream>>>(xw, src, dst, nrm, agg, E);
  k_finish<64, true><<<4096, 256, 0, stream>>>(agg, xw, dis, b3, (void*)h3, N);

  // ---- global max pool + head ----
  k_zero_f32<<<16, 256, 0, stream>>>(gmax, 64 * 64);
  k_pool<<<4096, 256, 0, stream>>>(h3, bat, gmax, N);
  k_head<<<1, 64, 0, stream>>>(gmax, Wh1, bh1, Wh2, bh2, (float*)d_out);
}
